// WordGCNPool_41652592836980
// MI455X (gfx1250) — matrix operand order repaired and checked by reading, not compiled, not run on hardware
//
#include <hip/hip_runtime.h>

#define N_WORDS 50000
#define N_DOCS  16384
#define DIM     256
#define N_EDGES 800000
#define N_XNZ   524288
#define ALPHA   0.7f
#define LN_EPS  1e-5f

typedef __attribute__((ext_vector_type(16))) __bf16        v16bf;
typedef __attribute__((ext_vector_type(8)))  float         v8f;
typedef __attribute__((ext_vector_type(4)))  unsigned int  u32x4;
typedef __attribute__((ext_vector_type(8)))  unsigned int  u32x8;

// round-to-nearest-even f32 -> bf16 bits (upper 16)
static __device__ __forceinline__ unsigned int rne16(float f) {
    unsigned int u = __builtin_bit_cast(unsigned int, f);
    u += ((u >> 16) & 1u) + 0x7FFFu;
    return u >> 16;
}
static __device__ __forceinline__ unsigned int pk2(float lo, float hi) {
    return (rne16(lo) & 0xFFFFu) | (rne16(hi) << 16);
}

// ---------------------------------------------------------------------------
// One-shot f32 -> bf16 pack (row-major preserved). 8 elements / thread.
// ---------------------------------------------------------------------------
__global__ __launch_bounds__(256) void f32_to_bf16_kernel(
    const float* __restrict__ in, unsigned int* __restrict__ out, int n8)
{
    int i = blockIdx.x * blockDim.x + threadIdx.x;   // index of 8-elem group
    if (i >= n8) return;
    const float4* p = reinterpret_cast<const float4*>(in) + (size_t)i * 2;
    float4 a = p[0], b = p[1];
    uint4 o;
    o.x = pk2(a.x, a.y);
    o.y = pk2(a.z, a.w);
    o.z = pk2(b.x, b.y);
    o.w = pk2(b.z, b.w);
    reinterpret_cast<uint4*>(out)[i] = o;
}

// ---------------------------------------------------------------------------
// SpMM via gather + native fp32 atomic scatter-add. One wave32 per nonzero;
// each lane: 8 contiguous floats (2x float4 gather, 8x global_atomic_add_f32).
// Gathered matrix (51.2MB) is L2-resident (192MB L2).
// ---------------------------------------------------------------------------
__global__ __launch_bounds__(256) void spmm_atomic_kernel(
    const int* __restrict__ rows, const int* __restrict__ cols,
    const float* __restrict__ vals, const float* __restrict__ H,
    float* __restrict__ out, int n_edges)
{
    int gid  = blockIdx.x * blockDim.x + threadIdx.x;
    int e    = gid >> 5;
    int lane = threadIdx.x & 31;
    if (e >= n_edges) return;
    int   r = rows[e];
    int   c = cols[e];
    float v = vals[e];
    const float4* src = reinterpret_cast<const float4*>(H + (size_t)c * DIM);
    float*        dst = out + (size_t)r * DIM + lane * 8;
    float4 x0 = src[lane * 2];
    float4 x1 = src[lane * 2 + 1];
    unsafeAtomicAdd(dst + 0, v * x0.x);
    unsafeAtomicAdd(dst + 1, v * x0.y);
    unsafeAtomicAdd(dst + 2, v * x0.z);
    unsafeAtomicAdd(dst + 3, v * x0.w);
    unsafeAtomicAdd(dst + 4, v * x1.x);
    unsafeAtomicAdd(dst + 5, v * x1.y);
    unsafeAtomicAdd(dst + 6, v * x1.z);
    unsafeAtomicAdd(dst + 7, v * x1.w);
}

// ---------------------------------------------------------------------------
// out[M,256] = act(A @ W^T [+ bias]) from pre-packed bf16 A [M,256] and
// bf16 W [256,256]. One wave32 per 16x64 output strip: one A fragment per
// K-step feeds 4 independent accumulators (4 n-tiles) -> 2.5 b128 loads per
// WMMA and 4 independent WMMA dependency chains (hides the bf16 WMMA->WMMA
// RAW hazard). K=256 in 8 steps, fully unrolled.
//   A (16-bit 16x32 layout): lane m=l&15; one contiguous 16-half chunk at
//     K = kk+hi*16.
//   B (32x16 layout): lane n=l&15; one contiguous 16-half chunk at K = kk+hi*16.
//   C/D: lane n=l&15; VGPR r -> M = r + hi*8.
// ---------------------------------------------------------------------------
template<bool RELU, bool BIAS>
__global__ __launch_bounds__(256) void gemm_bf16_wmma_kernel(
    const unsigned short* __restrict__ Abf,  // [M, 256] bf16
    const unsigned short* __restrict__ Wbf,  // [256, 256] bf16, out = A @ W^T
    const float* __restrict__ bias,          // [256] or nullptr
    float* __restrict__ out, int M)
{
    int lane  = threadIdx.x & 31;
    int strip = blockIdx.x * 8 + (threadIdx.x >> 5);
    int nstrips = M >> 2;                    // (M/16) m-tiles * 4 strips each
    if (strip >= nstrips) return;
    int m0 = (strip >> 2) << 4;
    int n0 = (strip & 3) << 6;               // 64-wide n strip
    int hi = lane >> 4;
    int lm = lane & 15;

    const unsigned short* arow  = Abf + (size_t)(m0 + lm) * DIM;
    const unsigned short* brow0 = Wbf + (size_t)(n0 +  0 + lm) * DIM + hi * 16;
    const unsigned short* brow1 = Wbf + (size_t)(n0 + 16 + lm) * DIM + hi * 16;
    const unsigned short* brow2 = Wbf + (size_t)(n0 + 32 + lm) * DIM + hi * 16;
    const unsigned short* brow3 = Wbf + (size_t)(n0 + 48 + lm) * DIM + hi * 16;

    v8f acc0 = {}, acc1 = {}, acc2 = {}, acc3 = {};
    #pragma unroll
    for (int kk = 0; kk < DIM; kk += 32) {
        u32x8 apk = *reinterpret_cast<const u32x8*>(arow + kk + hi * 16);
        v16bf af = __builtin_bit_cast(v16bf, apk);
        v16bf b0 = __builtin_bit_cast(v16bf, *reinterpret_cast<const u32x8*>(brow0 + kk));
        v16bf b1 = __builtin_bit_cast(v16bf, *reinterpret_cast<const u32x8*>(brow1 + kk));
        v16bf b2 = __builtin_bit_cast(v16bf, *reinterpret_cast<const u32x8*>(brow2 + kk));
        v16bf b3 = __builtin_bit_cast(v16bf, *reinterpret_cast<const u32x8*>(brow3 + kk));
        acc0 = __builtin_amdgcn_wmma_f32_16x16x32_bf16(false, af, false, b0, (short)0, acc0, false, false);
        acc1 = __builtin_amdgcn_wmma_f32_16x16x32_bf16(false, af, false, b1, (short)0, acc1, false, false);
        acc2 = __builtin_amdgcn_wmma_f32_16x16x32_bf16(false, af, false, b2, (short)0, acc2, false, false);
        acc3 = __builtin_amdgcn_wmma_f32_16x16x32_bf16(false, af, false, b3, (short)0, acc3, false, false);
    }

    #pragma unroll
    for (int j = 0; j < 4; ++j) {
        v8f acc = (j == 0) ? acc0 : (j == 1) ? acc1 : (j == 2) ? acc2 : acc3;
        int nc = n0 + j * 16 + lm;
        float bv = BIAS ? bias[nc] : 0.0f;
        #pragma unroll
        for (int r = 0; r < 8; ++r) {
            int m = m0 + r + (hi ? 8 : 0);
            float x = acc[r] + bv;
            if (RELU) x = fmaxf(x, 0.0f);
            out[(size_t)m * DIM + nc] = x;
        }
    }
}

// ---------------------------------------------------------------------------
// Fused: h = (1-a)*emb + a*hin ; LayerNorm(h)*g + b ; += emb  (doc-SpMM fold)
// One 256-thread block per row. wave32 shuffle + LDS reduction.
// ---------------------------------------------------------------------------
__global__ __launch_bounds__(256) void residual_ln_kernel(
    const float* __restrict__ emb, const float* __restrict__ hin,
    const float* __restrict__ g, const float* __restrict__ b,
    float* __restrict__ out)
{
    int row = blockIdx.x;
    int t   = threadIdx.x;
    float e = emb[(size_t)row * DIM + t];
    float x = (1.0f - ALPHA) * e + ALPHA * hin[(size_t)row * DIM + t];

    __shared__ float s1[8], s2[8];
    float v1 = x, v2 = x * x;
    #pragma unroll
    for (int o = 16; o > 0; o >>= 1) {
        v1 += __shfl_down(v1, o, 32);
        v2 += __shfl_down(v2, o, 32);
    }
    if ((t & 31) == 0) { s1[t >> 5] = v1; s2[t >> 5] = v2; }
    __syncthreads();
    float sum = 0.0f, sq = 0.0f;
    #pragma unroll
    for (int i = 0; i < 8; ++i) { sum += s1[i]; sq += s2[i]; }
    float mu  = sum * (1.0f / DIM);
    float var = sq * (1.0f / DIM) - mu * mu;
    float inv = rsqrtf(var + LN_EPS);
    out[(size_t)row * DIM + t] = (x - mu) * inv * g[t] + b[t] + e;
}

// ---------------------------------------------------------------------------
// Final classifier: out[m, 0..1] = doc[m,:] . cls_W[c,:] + cls_b[c]
// ---------------------------------------------------------------------------
__global__ __launch_bounds__(256) void cls_kernel(
    const float* __restrict__ doc, const float* __restrict__ Wc,
    const float* __restrict__ bc, float* __restrict__ out, int M)
{
    int m = blockIdx.x * blockDim.x + threadIdx.x;
    if (m >= M) return;
    const float4* dr = reinterpret_cast<const float4*>(doc + (size_t)m * DIM);
    const float4* w0 = reinterpret_cast<const float4*>(Wc);
    const float4* w1 = reinterpret_cast<const float4*>(Wc + DIM);
    float a0 = 0.0f, a1 = 0.0f;
    #pragma unroll 4
    for (int i = 0; i < DIM / 4; ++i) {
        float4 d = dr[i], x = w0[i], y = w1[i];
        a0 += d.x * x.x + d.y * x.y + d.z * x.z + d.w * x.w;
        a1 += d.x * y.x + d.y * y.y + d.z * y.z + d.w * y.w;
    }
    out[m * 2 + 0] = a0 + bc[0];
    out[m * 2 + 1] = a1 + bc[1];
}

extern "C" void kernel_launch(void* const* d_in, const int* in_sizes, int n_in,
                              void* d_out, int out_size, void* d_ws, size_t ws_size,
                              hipStream_t stream) {
    const int*   a_rows = (const int*)d_in[0];
    const int*   a_cols = (const int*)d_in[1];
    const float* a_vals = (const float*)d_in[2];
    const int*   x_rows = (const int*)d_in[3];
    const int*   x_cols = (const int*)d_in[4];
    const float* x_vals = (const float*)d_in[5];
    const float* emb    = (const float*)d_in[6];
    const float* W1     = (const float*)d_in[7];
    const float* W2     = (const float*)d_in[8];
    const float* ln_g   = (const float*)d_in[9];
    const float* ln_b   = (const float*)d_in[10];
    const float* mlp_W  = (const float*)d_in[11];
    const float* mlp_b  = (const float*)d_in[12];
    const float* cls_W  = (const float*)d_in[13];
    const float* cls_b  = (const float*)d_in[14];

    const size_t WORD_BYTES   = (size_t)N_WORDS * DIM * sizeof(float);  // 51,200,000
    const size_t DOC_BYTES    = (size_t)N_DOCS  * DIM * sizeof(float);  // 16,777,216
    const size_t WORDBF_BYTES = (size_t)N_WORDS * DIM * 2;              // 25,600,000
    const size_t WBF_BYTES    = (size_t)DIM * DIM * 2;                  // 131,072

    char* ws = (char*)d_ws;
    float*          t0   = (float*)(ws);
    float*          t1   = (float*)(ws + WORD_BYTES);
    unsigned short* abf  = (unsigned short*)(ws + 2 * WORD_BYTES);
    unsigned short* wbf  = (unsigned short*)(ws + 2 * WORD_BYTES + WORDBF_BYTES);
    float*          doc0 = (float*)(ws + 2 * WORD_BYTES + WORDBF_BYTES + WBF_BYTES);
    float*          doc1 = (float*)(ws + 2 * WORD_BYTES + WORDBF_BYTES + WBF_BYTES + DOC_BYTES);

    const int spmmA_blocks = (N_EDGES * 32 + 255) / 256;   // one wave per edge
    const int spmmX_blocks = (N_XNZ   * 32 + 255) / 256;
    const int gemmW_blocks = (N_WORDS / 4 + 7) / 8;        // 16x64 strips, 8 waves/blk
    const int gemmD_blocks = (N_DOCS  / 4 + 7) / 8;
    const int cvtW_blocks  = (N_WORDS * DIM / 8 + 255) / 256;
    const int cvtD_blocks  = (N_DOCS  * DIM / 8 + 255) / 256;
    const int cvtM_blocks  = (DIM * DIM / 8 + 255) / 256;

    // h = A @ emb
    hipMemsetAsync(t0, 0, WORD_BYTES, stream);
    spmm_atomic_kernel<<<spmmA_blocks, 256, 0, stream>>>(a_rows, a_cols, a_vals,
                                                         emb, t0, N_EDGES);
    // h = relu(h @ W1^T)   (bf16-pack h and W1 once, then pure-WMMA GEMM)
    f32_to_bf16_kernel<<<cvtW_blocks, 256, 0, stream>>>(t0, (unsigned int*)abf,
                                                        N_WORDS * DIM / 8);
    f32_to_bf16_kernel<<<cvtM_blocks, 256, 0, stream>>>(W1, (unsigned int*)wbf,
                                                        DIM * DIM / 8);
    gemm_bf16_wmma_kernel<true, false><<<gemmW_blocks, 256, 0, stream>>>(
        abf, wbf, nullptr, t1, N_WORDS);
    // h = A @ h
    hipMemsetAsync(t0, 0, WORD_BYTES, stream);
    spmm_atomic_kernel<<<spmmA_blocks, 256, 0, stream>>>(a_rows, a_cols, a_vals,
                                                         t1, t0, N_EDGES);
    // h = relu(h @ W2^T)
    f32_to_bf16_kernel<<<cvtW_blocks, 256, 0, stream>>>(t0, (unsigned int*)abf,
                                                        N_WORDS * DIM / 8);
    f32_to_bf16_kernel<<<cvtM_blocks, 256, 0, stream>>>(W2, (unsigned int*)wbf,
                                                        DIM * DIM / 8);
    gemm_bf16_wmma_kernel<true, false><<<gemmW_blocks, 256, 0, stream>>>(
        abf, wbf, nullptr, t1, N_WORDS);
    // word_plus = LN((1-a)*emb + a*h)*g + b + emb  (folds the two doc SpMMs)
    residual_ln_kernel<<<N_WORDS, 256, 0, stream>>>(emb, t1, ln_g, ln_b, t0);
    // doc = X @ word_plus
    hipMemsetAsync(doc0, 0, DOC_BYTES, stream);
    spmm_atomic_kernel<<<spmmX_blocks, 256, 0, stream>>>(x_rows, x_cols, x_vals,
                                                         t0, doc0, N_XNZ);
    // doc = relu(doc @ mlp_W^T + mlp_b)
    f32_to_bf16_kernel<<<cvtD_blocks, 256, 0, stream>>>(doc0, (unsigned int*)abf,
                                                        N_DOCS * DIM / 8);
    f32_to_bf16_kernel<<<cvtM_blocks, 256, 0, stream>>>(mlp_W, (unsigned int*)wbf,
                                                        DIM * DIM / 8);
    gemm_bf16_wmma_kernel<true, true><<<gemmD_blocks, 256, 0, stream>>>(
        abf, wbf, mlp_b, doc1, N_DOCS);
    // out = doc @ cls_W^T + cls_b
    cls_kernel<<<(N_DOCS + 255) / 256, 256, 0, stream>>>(doc1, cls_W, cls_b,
                                                         (float*)d_out, N_DOCS);
}
